// MAEEnhancedAttention_15547781611840
// MI455X (gfx1250) — matrix operand order repaired and checked by reading, not compiled, not run on hardware
//
#include <hip/hip_runtime.h>
#include <float.h>

// ---------------------------------------------------------------- constants
static constexpr int Bc   = 2;
static constexpr int S    = 2048;
static constexpr int SENC = 2048;
static constexpr int L    = 4096;      // SENC + S
static constexpr int HID  = 768;
static constexpr int H    = 12;
static constexpr int D    = 64;
static constexpr int NKV  = 2 * HID;   // 1536

typedef _Float16 h16;
typedef __attribute__((ext_vector_type(16))) _Float16 v16h;
typedef __attribute__((ext_vector_type(8)))  _Float16 v8h;
typedef __attribute__((ext_vector_type(8)))  float    v8f;

// ---------------------------------------------------------------- fragments
// A-matrix 16x32 f16 (M=lane&15): halves h -> K = (h<8?h:h+8) + 8*(lane>=16)
__device__ __forceinline__ v16h load_a_frag(const h16* base, int ld, int lane) {
  const h16* p = base + (size_t)(lane & 15) * ld + ((lane & 16) ? 8 : 0);
  v8h lo = *(const v8h*)p;
  v8h hi = *(const v8h*)(p + 16);
  v16h a;
#pragma unroll
  for (int i = 0; i < 8; ++i) { a[i] = lo[i]; a[i + 8] = hi[i]; }
  return a;
}

// B-matrix 32x16 f16 (N=lane&15): halves h -> K = (h<8?h:h+8) + 8*(lane>=16).
// 'base' points at (k0, n0) of a layout where each N-column is K-contiguous
// with stride 'ld' between columns.
__device__ __forceinline__ v16h load_b_frag(const h16* base, int ld, int lane) {
  const h16* p = base + (size_t)(lane & 15) * ld + ((lane & 16) ? 8 : 0);
  v8h lo = *(const v8h*)p;
  v8h hi = *(const v8h*)(p + 16);
  v16h b;
#pragma unroll
  for (int i = 0; i < 8; ++i) { b[i] = lo[i]; b[i + 8] = hi[i]; }
  return b;
}

__device__ __forceinline__ v8f wmma32(v16h a, v16h b, v8f c) {
  return __builtin_amdgcn_wmma_f32_16x16x32_f16(false, a, false, b, (short)0, c,
                                                false, false);
}

__device__ __forceinline__ float rmax16(float v) {
#pragma unroll
  for (int off = 1; off < 16; off <<= 1) v = fmaxf(v, __shfl_xor(v, off, 32));
  return v;
}
__device__ __forceinline__ float rsum16(float v) {
#pragma unroll
  for (int off = 1; off < 16; off <<= 1) v += __shfl_xor(v, off, 32);
  return v;
}

// ---------------------------------------------------------------- LayerNorm
__global__ void ln_q_kernel(const float* __restrict__ x,
                            const float* __restrict__ g,
                            const float* __restrict__ bb,
                            float* __restrict__ y32, h16* __restrict__ y16) {
  const int row = blockIdx.x;                 // 0 .. B*S-1
  const float* xr = x + (size_t)row * HID;
  const int tid = threadIdx.x;                // 256 threads
  float s = 0.f, s2 = 0.f;
  for (int i = tid; i < HID; i += 256) { float v = xr[i]; s += v; s2 += v * v; }
  __shared__ float sh[2][8];
  for (int off = 16; off > 0; off >>= 1) {
    s += __shfl_xor(s, off, 32); s2 += __shfl_xor(s2, off, 32);
  }
  if ((tid & 31) == 0) { sh[0][tid >> 5] = s; sh[1][tid >> 5] = s2; }
  __syncthreads();
  if (tid == 0) {
    float a = 0.f, c = 0.f;
    for (int i = 0; i < 8; ++i) { a += sh[0][i]; c += sh[1][i]; }
    sh[0][0] = a; sh[1][0] = c;
  }
  __syncthreads();
  const float mu  = sh[0][0] * (1.f / HID);
  const float var = sh[1][0] * (1.f / HID) - mu * mu;
  const float rs  = rsqrtf(var + 1e-12f);
  for (int i = tid; i < HID; i += 256) {
    float v = (xr[i] - mu) * rs * g[i] + bb[i];
    y32[(size_t)row * HID + i] = v;
    y16[(size_t)row * HID + i] = (h16)(v * 0.125f);   // fold 1/sqrt(D)
  }
}

__global__ void ln_kv_kernel(const float* __restrict__ x,
                             const float* __restrict__ g,
                             const float* __restrict__ bb,
                             h16* __restrict__ ekv) {
  const int row = blockIdx.x;                 // 0 .. B*S-1
  const int b = row / S, srow = row % S;
  const float* xr = x + (size_t)row * HID;
  const int tid = threadIdx.x;
  float s = 0.f, s2 = 0.f;
  for (int i = tid; i < HID; i += 256) { float v = xr[i]; s += v; s2 += v * v; }
  __shared__ float sh[2][8];
  for (int off = 16; off > 0; off >>= 1) {
    s += __shfl_xor(s, off, 32); s2 += __shfl_xor(s2, off, 32);
  }
  if ((tid & 31) == 0) { sh[0][tid >> 5] = s; sh[1][tid >> 5] = s2; }
  __syncthreads();
  if (tid == 0) {
    float a = 0.f, c = 0.f;
    for (int i = 0; i < 8; ++i) { a += sh[0][i]; c += sh[1][i]; }
    sh[0][0] = a; sh[1][0] = c;
  }
  __syncthreads();
  const float mu  = sh[0][0] * (1.f / HID);
  const float var = sh[1][0] * (1.f / HID) - mu * mu;
  const float rs  = rsqrtf(var + 1e-12f);
  h16* dst = ekv + ((size_t)b * L + SENC + srow) * HID;
  for (int i = tid; i < HID; i += 256)
    dst[i] = (h16)((xr[i] - mu) * rs * g[i] + bb[i]);
}

// ------------------------------------------------- f32 -> f16 conversions
__global__ void enc_to_ekv_kernel(const float* __restrict__ enc,
                                  h16* __restrict__ ekv) {
  const size_t total = (size_t)Bc * SENC * HID;
  const size_t stride = (size_t)gridDim.x * blockDim.x;
  for (size_t i = (size_t)blockIdx.x * blockDim.x + threadIdx.x; i < total; i += stride) {
    size_t b = i / ((size_t)SENC * HID);
    size_t rem = i - b * (size_t)SENC * HID;
    ekv[b * (size_t)L * HID + rem] = (h16)enc[i];
  }
}

__global__ void cvt_f16_kernel(const float* __restrict__ src,
                               h16* __restrict__ dst, int n) {
  int i = blockIdx.x * blockDim.x + threadIdx.x;
  if (i < n) dst[i] = (h16)src[i];
}

// -------------------------------------------------------- KV projection GEMM
// 16x64 output tile per wave: A-fragment reused across 4 B-tiles, with all
// five fragments of a k-step materialized before the WMMA burst so loads of
// step k+1 overlap the WMMAs of step k (no s_wait_loadcnt 0 serialization).
// out[gm, o] = sum_i ekv[gm, i] * Wkv[o, i] + bias[o]
// o < 768 -> K cache [B,H,L,D] ; o >= 768 -> V cache transposed [B,H,D,L]
__global__ void kv_proj_kernel(const h16* __restrict__ ekv,
                               const h16* __restrict__ wkv,
                               const float* __restrict__ bias,
                               h16* __restrict__ Kc, h16* __restrict__ Vt) {
  const int lane = threadIdx.x & 31;
  const int wv   = threadIdx.x >> 5;
  const int m0   = (blockIdx.x * 8 + wv) * 16;
  const int n0   = blockIdx.y * 64;
  const h16* abase = ekv + (size_t)m0 * HID;
  const h16* bbase = wkv + (size_t)n0 * HID; // Wkv row o is K-contiguous
  v8f acc[4] = {v8f{}, v8f{}, v8f{}, v8f{}};
#pragma unroll 2
  for (int k0 = 0; k0 < HID; k0 += 32) {
    v16h a = load_a_frag(abase + k0, HID, lane);
    v16h bf[4];
#pragma unroll
    for (int j = 0; j < 4; ++j)
      bf[j] = load_b_frag(bbase + (size_t)j * 16 * HID + k0, HID, lane);
#pragma unroll
    for (int j = 0; j < 4; ++j)
      acc[j] = wmma32(a, bf[j], acc[j]);
  }
  // n0 is 64-aligned -> whole supertile is one (t, head) pair
  const int t  = n0 / HID;
  const int hd = (n0 % HID) / D;
  const int hi = (lane >> 4) & 1;
#pragma unroll
  for (int j = 0; j < 4; ++j) {
    const int d  = j * 16 + (lane & 15);
    const float bv = bias[n0 + j * 16 + (lane & 15)];
#pragma unroll
    for (int r = 0; r < 8; ++r) {
      const int gm = m0 + r + 8 * hi;        // gm = b*L + l
      const int b_ = gm >> 12;               // L = 4096
      const int l  = gm & (L - 1);
      const float val = acc[j][r] + bv;
      if (t == 0)
        Kc[(((size_t)b_ * H + hd) * L + l) * D + d] = (h16)val;
      else
        Vt[(((size_t)b_ * H + hd) * D + d) * L + l] = (h16)val;
    }
  }
}

// ----------------------------------------------------------- flash attention
__global__ void flash_attn_kernel(const h16* __restrict__ qf,
                                  const h16* __restrict__ Kc,
                                  const h16* __restrict__ Vt,
                                  const unsigned char* __restrict__ amask,
                                  const unsigned char* __restrict__ dmask,
                                  h16* __restrict__ outf) {
  __shared__ __align__(16) h16 pbuf[4][16][32];
  const int lane = threadIdx.x & 31;
  const int wv   = threadIdx.x >> 5;         // 4 waves / block
  const int bh   = blockIdx.y;               // b*H + hd
  const int b    = bh / H;
  const int hd   = bh % H;
  const int s0   = (blockIdx.x * 4 + wv) * 16;
  const int hi   = (lane >> 4) & 1;
  const int n    = lane & 15;

  // Q fragments (hoisted; q already scaled by 1/sqrt(D))
  const h16* qbase = qf + ((size_t)b * S + s0) * HID + hd * D;
  const v16h aq0 = load_a_frag(qbase,      HID, lane);   // d = 0..31
  const v16h aq1 = load_a_frag(qbase + 32, HID, lane);   // d = 32..63

  const h16* KH = Kc + (size_t)bh * L * D;   // [l][d], d-contiguous
  const h16* VH = Vt + (size_t)bh * D * L;   // [d][l], l-contiguous

  v8f acc[4] = {v8f{}, v8f{}, v8f{}, v8f{}};
  float mrun[8], lsum[8];
#pragma unroll
  for (int r = 0; r < 8; ++r) { mrun[r] = -3.0e38f; lsum[r] = 0.f; }

  const unsigned char* am = amask + (size_t)b * S;
  const unsigned char* dm = dmask + (size_t)b * S * S;

  for (int l0 = 0; l0 < L; l0 += 32) {
    // ---- issue ALL VMEM for this chunk up front: 4 K-frags + 4 V-frags.
    // The V fragments are consumed only after the softmax VALU section, so
    // they have ~150 instructions + a barrier of latency-hiding slack.
    v16h bk[4], bv[4];
    bk[0] = load_b_frag(KH + (size_t)l0 * D,            D, lane);
    bk[1] = load_b_frag(KH + (size_t)l0 * D + 32,       D, lane);
    bk[2] = load_b_frag(KH + (size_t)(l0 + 16) * D,      D, lane);
    bk[3] = load_b_frag(KH + (size_t)(l0 + 16) * D + 32, D, lane);
#pragma unroll
    for (int t = 0; t < 4; ++t)
      bv[t] = load_b_frag(VH + (size_t)(t * 16) * L + l0, L, lane);
    // ---- prefetch next chunk of K into L2/WGP$ while this chunk computes
    if (l0 + 32 < L)
      __builtin_prefetch(KH + (size_t)(l0 + 32) * D + lane * 8, 0, 0);

    // ---- scores for 32 columns (two 16x16 C tiles)
    v8f cs0 = {}, cs1 = {};
    cs0 = wmma32(aq0, bk[0], cs0);
    cs0 = wmma32(aq1, bk[1], cs0);
    cs1 = wmma32(aq0, bk[2], cs1);
    cs1 = wmma32(aq1, bk[3], cs1);

    // ---- mask (only decoder half of L)
    if (l0 >= SENC) {
      const int j0 = l0 - SENC + n;
      const int j1 = j0 + 16;
      const bool a0 = am[j0] != 0, a1 = am[j1] != 0;
#pragma unroll
      for (int r = 0; r < 8; ++r) {
        const int srow = s0 + r + 8 * hi;
        const unsigned char* dr = dm + (size_t)srow * S;
        if (!(a0 && dr[j0] && (srow != j0))) cs0[r] = -3.0e38f;
        if (!(a1 && dr[j1] && (srow != j1))) cs1[r] = -3.0e38f;
      }
    }
    // ---- online softmax + stage P tile (f32 C-layout -> f16 in LDS)
#pragma unroll
    for (int r = 0; r < 8; ++r) {
      float mt = rmax16(fmaxf(cs0[r], cs1[r]));
      float mn = fmaxf(mrun[r], mt);
      float sc = __expf(mrun[r] - mn);
      mrun[r] = mn;
      float p0 = __expf(cs0[r] - mn);
      float p1 = __expf(cs1[r] - mn);
      lsum[r] = lsum[r] * sc + rsum16(p0 + p1);
      acc[0][r] *= sc; acc[1][r] *= sc; acc[2][r] *= sc; acc[3][r] *= sc;
      pbuf[wv][r + 8 * hi][n]      = (h16)p0;
      pbuf[wv][r + 8 * hi][16 + n] = (h16)p1;
    }
    __syncthreads();                         // LDS visibility (uniform trip count)
    // ---- reload P as 16x32 A-fragment
    v16h pa;
    {
      const h16* pr = &pbuf[wv][lane & 15][(lane & 16) ? 8 : 0];
      v8h lo = *(const v8h*)pr;
      v8h hh = *(const v8h*)(pr + 16);
#pragma unroll
      for (int i = 0; i < 8; ++i) { pa[i] = lo[i]; pa[i + 8] = hh[i]; }
    }
    // ---- P @ V  (D = 4 N-tiles of 16); V fragments already resident
#pragma unroll
    for (int t = 0; t < 4; ++t)
      acc[t] = wmma32(pa, bv[t], acc[t]);
    __syncthreads();
  }
  // ---- epilogue: normalize, write f16 attention output [B,S,HID]
#pragma unroll
  for (int t = 0; t < 4; ++t)
#pragma unroll
    for (int r = 0; r < 8; ++r) {
      const int srow = s0 + r + 8 * hi;
      const int d = t * 16 + n;
      outf[((size_t)b * S + srow) * HID + hd * D + d] = (h16)(acc[t][r] / lsum[r]);
    }
}

// -------------------------------------------- dense GEMM + bias + residual
// 16x64 output tile per wave; grid: x = (B*S/16)/8, y = HID/64
__global__ void dense_kernel(const h16* __restrict__ a16,
                             const h16* __restrict__ w16,
                             const float* __restrict__ bias,
                             const float* __restrict__ resid,
                             float* __restrict__ out) {
  const int lane = threadIdx.x & 31;
  const int wv   = threadIdx.x >> 5;
  const int m0   = (blockIdx.x * 8 + wv) * 16;
  const int n0   = blockIdx.y * 64;
  const h16* abase = a16 + (size_t)m0 * HID;
  const h16* bbase = w16 + (size_t)n0 * HID;
  v8f acc[4] = {v8f{}, v8f{}, v8f{}, v8f{}};
#pragma unroll 2
  for (int k0 = 0; k0 < HID; k0 += 32) {
    v16h a = load_a_frag(abase + k0, HID, lane);
    v16h bf[4];
#pragma unroll
    for (int j = 0; j < 4; ++j)
      bf[j] = load_b_frag(bbase + (size_t)j * 16 * HID + k0, HID, lane);
#pragma unroll
    for (int j = 0; j < 4; ++j)
      acc[j] = wmma32(a, bf[j], acc[j]);
  }
  const int hi = (lane >> 4) & 1;
#pragma unroll
  for (int j = 0; j < 4; ++j) {
    const int nn = n0 + j * 16 + (lane & 15);
    const float bv = bias[nn];
#pragma unroll
    for (int r = 0; r < 8; ++r) {
      const size_t gm = (size_t)m0 + r + 8 * hi;
      out[gm * HID + nn] = acc[j][r] + bv + resid[gm * HID + nn];
    }
  }
}

// ---------------------------------------------------------------- launcher
extern "C" void kernel_launch(void* const* d_in, const int* in_sizes, int n_in,
                              void* d_out, int out_size, void* d_ws, size_t ws_size,
                              hipStream_t stream) {
  const float* qx   = (const float*)d_in[0];
  const float* kvx  = (const float*)d_in[1];
  const float* enc  = (const float*)d_in[2];
  const unsigned char* amask = (const unsigned char*)d_in[3];
  const unsigned char* dmask = (const unsigned char*)d_in[4];
  const float* wkvw = (const float*)d_in[7];
  const float* wkvb = (const float*)d_in[8];
  const float* dw   = (const float*)d_in[9];
  const float* db   = (const float*)d_in[10];
  const float* ng   = (const float*)d_in[11];
  const float* nb   = (const float*)d_in[12];
  float* out = (float*)d_out;

  // workspace carve-up (256B aligned regions)
  char* ws = (char*)d_ws;
  size_t off = 0;
  auto carve = [&](size_t bytes) { void* p = ws + off; off += (bytes + 255) & ~(size_t)255; return p; };
  float* q32   = (float*)carve((size_t)Bc * S * HID * 4);
  h16*   q16   = (h16*)  carve((size_t)Bc * S * HID * 2);
  h16*   ekv   = (h16*)  carve((size_t)Bc * L * HID * 2);
  h16*   Kc    = (h16*)  carve((size_t)Bc * H * L * D * 2);
  h16*   Vt    = (h16*)  carve((size_t)Bc * H * D * L * 2);
  h16*   att16 = (h16*)  carve((size_t)Bc * S * HID * 2);
  h16*   wkv16 = (h16*)  carve((size_t)NKV * HID * 2);
  h16*   dw16  = (h16*)  carve((size_t)HID * HID * 2);

  // 1) LayerNorms + f16 staging
  ln_q_kernel<<<Bc * S, 256, 0, stream>>>(qx, ng, nb, q32, q16);
  ln_kv_kernel<<<Bc * S, 256, 0, stream>>>(kvx, ng, nb, ekv);
  enc_to_ekv_kernel<<<4096, 256, 0, stream>>>(enc, ekv);
  cvt_f16_kernel<<<(NKV * HID + 255) / 256, 256, 0, stream>>>(wkvw, wkv16, NKV * HID);
  cvt_f16_kernel<<<(HID * HID + 255) / 256, 256, 0, stream>>>(dw, dw16, HID * HID);

  // 2) KV projection -> K cache [B,H,L,D], V cache [B,H,D,L]
  {
    dim3 grid((Bc * L / 16) / 8, NKV / 64);
    kv_proj_kernel<<<grid, 256, 0, stream>>>(ekv, wkv16, wkvb, Kc, Vt);
  }

  // 3) flash attention
  {
    dim3 fgrid(S / 16 / 4, Bc * H);
    flash_attn_kernel<<<fgrid, 128, 0, stream>>>(q16, Kc, Vt, amask, dmask, att16);
  }

  // 4) dense + bias + residual (residual = pre-normed q_hs f32)
  {
    dim3 grid((Bc * S / 16) / 8, HID / 64);
    dense_kernel<<<grid, 256, 0, stream>>>(att16, dw16, db, q32, out);
  }
}
